// SimpleCNN_82798379532474
// MI455X (gfx1250) — matrix operand order repaired and checked by reading, not compiled, not run on hardware
//
#include <hip/hip_runtime.h>

// ---------------------------------------------------------------------------
// CDNA5 / gfx1250 types for WMMA (wave32)
// ---------------------------------------------------------------------------
typedef __attribute__((ext_vector_type(16))) __bf16 v16bf;
typedef __attribute__((ext_vector_type(8)))  __bf16 v8bf;
typedef __attribute__((ext_vector_type(8)))  float  v8f;

union AFrag { v16bf v; v8bf h[2]; };

#define WMMA_BF16(a, b, c) \
    __builtin_amdgcn_wmma_f32_16x16x32_bf16(false, (a), false, (b), (short)0, (c), false, false)

// Problem constants
#define NBATCH   4096
#define CONV2_NGROUPS 12544   // (4096*196/16) M-tiles / 4 waves per block

// ---------------------------------------------------------------------------
// Kernel 0: build dense DCLS 5x5 kernels from (w, p) via bilinear scatter.
// One thread per (co, ci) pair; scatter into a local 6x6 grid, trim to 5x5.
// outF (optional): fp32, layout [co*ci][25]        (for conv1 VALU path)
// outB (optional): bf16, layout [tap][co][ci]       (for conv2 WMMA B-frags)
// ---------------------------------------------------------------------------
__global__ void build_dcls(const float* __restrict__ w, const float* __restrict__ p,
                           float* outF, __bf16* outB, int Co, int Ci, int Kc) {
    int idx = blockIdx.x * blockDim.x + threadIdx.x;
    if (idx >= Co * Ci) return;
    int co = idx / Ci, ci = idx - co * Ci;
    float acc[36];
    for (int i = 0; i < 36; ++i) acc[i] = 0.f;
    int CiKc = Ci * Kc;
    for (int q = 0; q < Kc; ++q) {
        float wq = w[(co * Ci + ci) * Kc + q];
        float pa = p[(co * Ci + ci) * Kc + q];              // dim 0
        float pb = p[Co * CiKc + (co * Ci + ci) * Kc + q];  // dim 1
        pa = fminf(fmaxf(pa, -2.f), 2.f) + 2.f;             // [0, 4]
        pb = fminf(fmaxf(pb, -2.f), 2.f) + 2.f;
        int i1 = (int)floorf(pa), i2 = (int)floorf(pb);
        float r1 = pa - (float)i1, r2 = pb - (float)i2;
        acc[i1 * 6 + i2]           += wq * (1.f - r1) * (1.f - r2);
        acc[(i1 + 1) * 6 + i2]     += wq * r1 * (1.f - r2);
        acc[i1 * 6 + i2 + 1]       += wq * (1.f - r1) * r2;
        acc[(i1 + 1) * 6 + i2 + 1] += wq * r1 * r2;
    }
    for (int kh = 0; kh < 5; ++kh)
        for (int kw = 0; kw < 5; ++kw) {
            float v = acc[kh * 6 + kw];
            if (outF) outF[idx * 25 + kh * 5 + kw] = v;
            if (outB) outB[((kh * 5 + kw) * Co + co) * Ci + ci] = (__bf16)v;
        }
}

// ---------------------------------------------------------------------------
// Kernel 0b: fp32 -> bf16 conversion (fc1 weights)
// ---------------------------------------------------------------------------
__global__ void cvt_bf16(const float* __restrict__ in, __bf16* __restrict__ out, int n) {
    int i = blockIdx.x * blockDim.x + threadIdx.x;
    if (i < n) out[i] = (__bf16)in[i];
}

// ---------------------------------------------------------------------------
// Kernel 1: conv1 (Cin=1) + bias + relu + maxpool2, fused. One block = 1 image.
// Image (28x28) in zero-padded 32x32 LDS tile; 5x5 taps for 32 couts in LDS.
// Output act1: bf16, channel-last (b, y, x, c) for conv2 A-fragment loads.
// ---------------------------------------------------------------------------
__global__ __launch_bounds__(256) void conv1_fused(const float* __restrict__ x,
                                                   const float* __restrict__ k1f,
                                                   const float* __restrict__ b1,
                                                   __bf16* __restrict__ act1) {
    __shared__ float img[32 * 32];
    __shared__ float k1s[32 * 25];
    __shared__ float b1s[32];
    int tid = threadIdx.x;
    int b = blockIdx.x;
    for (int i = tid; i < 1024; i += 256) img[i] = 0.f;
    __syncthreads();
    for (int i = tid; i < 784; i += 256) {
        int y = i / 28, xx = i - 28 * y;
        img[(y + 2) * 32 + (xx + 2)] = x[b * 784 + i];
    }
    for (int i = tid; i < 800; i += 256) k1s[i] = k1f[i];
    if (tid < 32) b1s[tid] = b1[tid];
    __syncthreads();
    if (tid < 196) {
        int py = tid / 14, px = tid - 14 * py;
        __bf16* optr = act1 + ((b * 14 + py) * 14 + px) * 32;
        for (int c = 0; c < 32; ++c) {
            const float* kk = k1s + c * 25;
            float m = -3.4e38f;
#pragma unroll
            for (int s = 0; s < 4; ++s) {              // 2x2 pool window
                int oy = 2 * py + (s >> 1), ox = 2 * px + (s & 1);
                float sum = 0.f;
#pragma unroll
                for (int t = 0; t < 25; ++t) {
                    int kh = t / 5, kw = t - 5 * (t / 5);
                    sum += img[(oy + kh) * 32 + ox + kw] * kk[t];
                }
                m = fmaxf(m, sum);
            }
            optr[c] = (__bf16)fmaxf(m + b1s[c], 0.f);  // pool(relu(v+b)) == relu(max+b)
        }
    }
}

// ---------------------------------------------------------------------------
// Kernel 2: conv2 as implicit GEMM on v_wmma_f32_16x16x32_bf16, fused with
// bias + maxpool2 + relu. M = B*196 output pixels (pool-window-major order:
// p = 4*win + sub), N = couts (this block: 32 of 64, selected by blockIdx.y),
// K = 25 taps x 32 channels -> one WMMA K-step per tap.
// Dense K2 half (25x32x32 bf16 = 50 KB) staged in LDS once per block.
// Pooled output written bf16 in FC-flatten order: act2[b*3136 + c*49 + win].
// ---------------------------------------------------------------------------
__global__ __launch_bounds__(128) void conv2_wmma(const __bf16* __restrict__ act1,
                                                  const __bf16* __restrict__ k2b,
                                                  const float* __restrict__ b2,
                                                  __bf16* __restrict__ act2) {
    __shared__ __bf16 k2s[25 * 32 * 32];   // [tap][cout_local][cin]  (50 KB)
    const int half = blockIdx.y;           // 0 or 1: which 32 couts
    {   // stage: per tap, 1024 contiguous bf16 from global offset half*1024
        const uint4* src = (const uint4*)k2b;
        uint4* dst = (uint4*)k2s;
        for (int i = threadIdx.x; i < 3200; i += 128) {   // 3200 * 16B = 50 KB
            int t = i >> 7, j = i & 127;
            dst[i] = src[t * 256 + half * 128 + j];
        }
    }
    __syncthreads();

    const int lane = threadIdx.x & 31;
    const int wv   = threadIdx.x >> 5;
    const int mrow = lane & 15;
    const bool hi  = lane >= 16;
    const int choff = hi ? 16 : 0;         // A-frag channel offset (contiguous K, matches B)
    const int koff  = hi ? 16 : 0;         // B-frag K offset (contiguous K, matches A)

    for (int g = blockIdx.x; g < CONV2_NGROUPS; g += gridDim.x) {
        const int g0 = (g * 4 + wv) * 16;  // this wave's 16 output pixels
        const int gi = g0 + mrow;          // this lane's pixel row (M)
        const int b  = gi / 196;
        const int p  = gi - b * 196;
        const int win = p >> 2, sub = p & 3;
        const int py = win / 7, px = win - py * 7;
        const int oh = 2 * py + (sub >> 1), ow = 2 * px + (sub & 1);

        v8f c0 = {}, c1 = {};
        for (int t = 0; t < 25; ++t) {
            const int kh = t / 5, kw = t - 5 * (t / 5);
            const int ih = oh + kh - 2, iw = ow + kw - 2;
            AFrag a;
            if ((unsigned)ih < 14u && (unsigned)iw < 14u) {
                const __bf16* ap = act1 + (((b * 14 + ih) * 14 + iw) * 32 + choff);
                a.h[0] = *(const v8bf*)ap;          // K {0..7}/{16..23}
                a.h[1] = *(const v8bf*)(ap + 8);    // K {8..15}/{24..31}
            } else {
                v16bf z = {};
                a.v = z;                             // zero padding
            }
            const __bf16* bp = k2s + (t * 32 + mrow) * 32 + koff;
            v16bf bf0 = *(const v16bf*)bp;                 // cout_local = mrow
            v16bf bf1 = *(const v16bf*)(bp + 16 * 32);     // cout_local = 16+mrow
            c0 = WMMA_BF16(a.v, bf0, c0);
            c1 = WMMA_BF16(a.v, bf1, c1);
        }

        // bias + 2x2 maxpool (accumulator rows 4k..4k+3 = one window) + relu
        const int wg0 = (g0 >> 2) + (hi ? 2 : 0);   // pool-window ids for this lane
#pragma unroll
        for (int n = 0; n < 2; ++n) {
            v8f c = (n == 0) ? c0 : c1;
            const int col = half * 32 + n * 16 + mrow;    // global cout
            const float bias = b2[col];
            const float pA = fmaxf(fmaxf(c[0], c[1]), fmaxf(c[2], c[3])) + bias;
            const float pB = fmaxf(fmaxf(c[4], c[5]), fmaxf(c[6], c[7])) + bias;
            int wgA = wg0, wgB = wg0 + 1;
            int bbA = wgA / 49, wnA = wgA - bbA * 49;
            int bbB = wgB / 49, wnB = wgB - bbB * 49;
            act2[bbA * 3136 + col * 49 + wnA] = (__bf16)fmaxf(pA, 0.f);
            act2[bbB * 3136 + col * 49 + wnB] = (__bf16)fmaxf(pB, 0.f);
        }
    }
}

// ---------------------------------------------------------------------------
// Kernel 3: FC1 (WMMA GEMM, M=4096 N=128 K=3136) + bias + relu, fused with
// FC2 (N=10) through a 64x128 LDS h-buffer. Block = 4 waves = 64 batch rows.
// fc1_w bf16 K-chunk [128][32] staged through LDS each iteration.
// ---------------------------------------------------------------------------
__global__ __launch_bounds__(128) void fc_fused(const __bf16* __restrict__ act2,
                                                const __bf16* __restrict__ fc1wb,
                                                const float* __restrict__ fc1_b,
                                                const float* __restrict__ fc2_w,
                                                const float* __restrict__ fc2_b,
                                                float* __restrict__ out) {
    __shared__ __bf16 stage[128 * 32];   // fc1_w chunk [n][k]   (8 KB)
    __shared__ float  hbuf[64 * 128];    // relu(fc1) tile       (32 KB)

    const int lane = threadIdx.x & 31;
    const int wv   = threadIdx.x >> 5;
    const int mrow = lane & 15;
    const bool hi  = lane >= 16;
    const int base = blockIdx.x * 64;
    const int arow = base + wv * 16 + mrow;
    const __bf16* aptr = act2 + (size_t)arow * 3136 + (hi ? 16 : 0);

    v8f c[8];
#pragma unroll
    for (int n = 0; n < 8; ++n) { v8f z = {}; c[n] = z; }

    const uint4* wsrc = (const uint4*)(fc1wb + (size_t)threadIdx.x * 3136);
    uint4* wdst = (uint4*)(stage + threadIdx.x * 32);

    for (int kc = 0; kc < 98; ++kc) {
        const int k0 = kc * 32;
        // stage fc1_w[:, k0:k0+32]: each thread copies its row's 64 bytes
        wdst[0] = wsrc[kc * 4 + 0];
        wdst[1] = wsrc[kc * 4 + 1];
        wdst[2] = wsrc[kc * 4 + 2];
        wdst[3] = wsrc[kc * 4 + 3];
        __builtin_prefetch(wsrc + kc * 4 + 4, 0, 1);   // global_prefetch next chunk
        __syncthreads();

        AFrag a;
        const __bf16* ap = aptr + k0;
        a.h[0] = *(const v8bf*)ap;
        a.h[1] = *(const v8bf*)(ap + 8);
#pragma unroll
        for (int n = 0; n < 8; ++n) {
            const __bf16* bp = stage + (n * 16 + mrow) * 32 + (hi ? 16 : 0);
            v16bf bf = *(const v16bf*)bp;
            c[n] = WMMA_BF16(a.v, bf, c[n]);
        }
        __syncthreads();   // before next stage overwrite
    }

    // bias + relu -> LDS h-buffer (C layout: VGPR r = row r / 8+r)
#pragma unroll
    for (int n = 0; n < 8; ++n) {
        const int col = n * 16 + mrow;
        const float bias = fc1_b[col];
#pragma unroll
        for (int r = 0; r < 8; ++r) {
            const int m = (hi ? 8 : 0) + r;
            hbuf[(wv * 16 + m) * 128 + col] = fmaxf(c[n][r] + bias, 0.f);
        }
    }
    __syncthreads();

    // FC2: 64 rows x 10 outputs per block
    for (int idx = threadIdx.x; idx < 640; idx += 128) {
        const int row = idx / 10, j = idx - row * 10;
        float s = fc2_b[j];
        const float* wrow = fc2_w + j * 128;
        const float* hrow = hbuf + row * 128;
#pragma unroll 8
        for (int n = 0; n < 128; ++n) s += hrow[n] * wrow[n];
        out[(base + row) * 10 + j] = s;
    }
}

// ---------------------------------------------------------------------------
// Launcher
// Inputs: 0:x 1:w1 2:p1 3:b1 4:w2 5:p2 6:b2 7:fc1_w 8:fc1_b 9:fc2_w 10:fc2_b
// Workspace layout (bytes):
//   k1f   @ 0        : 32*25*4        = 3200
//   k2b   @ 4096     : 25*64*32*2     = 102400
//   fc1wb @ 106496   : 128*3136*2     = 802816
//   act1  @ 909312   : 4096*196*32*2  = 51380224   (b,y,x,c) bf16
//   act2  @ 52289536 : 4096*3136*2    = 25690112   (b, c*49+win) bf16
// Total ~78 MB.
// ---------------------------------------------------------------------------
extern "C" void kernel_launch(void* const* d_in, const int* in_sizes, int n_in,
                              void* d_out, int out_size, void* d_ws, size_t ws_size,
                              hipStream_t stream) {
    const float* x     = (const float*)d_in[0];
    const float* w1    = (const float*)d_in[1];
    const float* p1    = (const float*)d_in[2];
    const float* b1    = (const float*)d_in[3];
    const float* w2    = (const float*)d_in[4];
    const float* p2    = (const float*)d_in[5];
    const float* b2    = (const float*)d_in[6];
    const float* fc1w  = (const float*)d_in[7];
    const float* fc1b  = (const float*)d_in[8];
    const float* fc2w  = (const float*)d_in[9];
    const float* fc2b  = (const float*)d_in[10];
    float* out = (float*)d_out;

    char* ws = (char*)d_ws;
    float*  k1f   = (float*)(ws + 0);
    __bf16* k2b   = (__bf16*)(ws + 4096);
    __bf16* fc1wb = (__bf16*)(ws + 106496);
    __bf16* act1  = (__bf16*)(ws + 909312);
    __bf16* act2  = (__bf16*)(ws + 52289536);

    build_dcls<<<1, 32, 0, stream>>>(w1, p1, k1f, nullptr, 32, 1, 16);
    build_dcls<<<8, 256, 0, stream>>>(w2, p2, nullptr, k2b, 64, 32, 32);
    cvt_bf16<<<1568, 256, 0, stream>>>(fc1w, fc1wb, 128 * 3136);
    conv1_fused<<<NBATCH, 256, 0, stream>>>(x, k1f, b1, act1);
    conv2_wmma<<<dim3(512, 2), 128, 0, stream>>>(act1, k2b, b2, act2);
    fc_fused<<<NBATCH / 64, 128, 0, stream>>>(act2, fc1wb, fc1b, fc2w, fc2b, out);
}
